// FCNNaccBaseline_21620865368616
// MI455X (gfx1250) — matrix operand order repaired and checked by reading, not compiled, not run on hardware
//
#include <hip/hip_runtime.h>
#include <hip/hip_bf16.h>
#include <math.h>

// ---------------------------------------------------------------------------
// FCN + NALU pipeline for MI455X (gfx1250, wave32, WMMA f16 16x16x32).
//
//   conv1(1->128,k8) +BN+ReLU -> conv2(128->256,k5)+BN+ReLU
//   -> conv3(256->128,k3)+BN+ReLU -> mean over length -> NALU(128->128)
//   -> NALU(128->16) -> linear(16->1)
//
// B=128, Lin=4096, Lout=4097 for all convs (pad = k/2 with stride 1).
// Conv biases cancel exactly through batch-stat BN and are omitted.
// Activations stored position-major [b][pos][ch] in f16.
// ---------------------------------------------------------------------------

typedef __attribute__((ext_vector_type(16))) _Float16 v16h;
typedef __attribute__((ext_vector_type(8)))  _Float16 v8h;
typedef __attribute__((ext_vector_type(4)))  _Float16 v4h;
typedef __attribute__((ext_vector_type(8)))  float    v8f;

#define B_    128
#define LX    4096
#define LOUT  4097
#define INV_N (1.0f / (128.0f * 4097.0f))

// ------------------------------ weight packing -----------------------------
// A-fragment element order (16-bit A 16x32, ISA 7.12.2):
//   lane = h*16 + m  (h = half, m = row within subtile)
//   VGPR v packs K = kf(v,h) , kf+1 with kf = (v/4)*16 + h*8 + (v%4)*2
// Packed layout: wp[(((kk*CH + ch)*MT + mt)*32 + lane)*16 + e]

__global__ void pack_w2_kernel(const float* __restrict__ w2, _Float16* __restrict__ wp) {
  int idx = blockIdx.x * 256 + threadIdx.x;
  if (idx >= 5 * 4 * 16 * 32 * 16) return;
  int e = idx & 15, lane = (idx >> 4) & 31, mt = (idx >> 9) & 15;
  int ch = (idx >> 13) & 3, kk = idx >> 15;
  int v = e >> 1, half = lane >> 4;
  int kf = (v >> 2) * 16 + half * 8 + (v & 3) * 2 + (e & 1);
  int ci = ch * 32 + kf;
  int co = mt * 16 + (lane & 15);
  wp[idx] = (_Float16)w2[(co * 128 + ci) * 5 + kk];   // w2: (256,128,5)
}

__global__ void pack_w3_kernel(const float* __restrict__ w3, _Float16* __restrict__ wp) {
  int idx = blockIdx.x * 256 + threadIdx.x;
  if (idx >= 3 * 8 * 8 * 32 * 16) return;
  int e = idx & 15, lane = (idx >> 4) & 31, mt = (idx >> 9) & 7;
  int ch = (idx >> 12) & 7, kk = idx >> 15;
  int v = e >> 1, half = lane >> 4;
  int kf = (v >> 2) * 16 + half * 8 + (v & 3) * 2 + (e & 1);
  int ci = ch * 32 + kf;
  int co = mt * 16 + (lane & 15);
  wp[idx] = (_Float16)w3[(co * 256 + ci) * 3 + kk];   // w3: (128,256,3)
}

// ------------------------------ conv1 (in_ch = 1) ---------------------------
// Direct conv, writes y1t[b][p][co] f16 (pre-BN) + per-channel sum/sumsq.
__global__ void __launch_bounds__(256) conv1_kernel(
    const float* __restrict__ x, const float* __restrict__ w1,
    _Float16* __restrict__ y1t, float* __restrict__ stats1) {
  const int tile = blockIdx.x;          // 33 tiles of 128 positions
  const int b    = blockIdx.y;
  const int tid  = threadIdx.x;
  __shared__ float xt[135];             // positions p0-4 .. p0+130
  __shared__ float wsh[128 * 8];
  __shared__ float lsum[128], lssq[128];
  const int p0 = tile * 128;
  for (int i = tid; i < 135; i += 256) {
    int p = p0 - 4 + i;
    xt[i] = (p >= 0 && p < LX) ? x[(size_t)b * LX + p] : 0.0f;
  }
  for (int i = tid; i < 1024; i += 256) wsh[i] = w1[i];
  if (tid < 128) { lsum[tid] = 0.f; lssq[tid] = 0.f; }
  __syncthreads();
  const int co = tid & 127, row = tid >> 7;
  float s = 0.f, q = 0.f;
  for (int pp = row; pp < 128; pp += 2) {
    int p = p0 + pp;
    if (p < LOUT) {
      float acc = 0.f;
#pragma unroll
      for (int k = 0; k < 8; k++) acc += wsh[co * 8 + k] * xt[pp + k];
      y1t[((size_t)b * LOUT + p) * 128 + co] = (_Float16)acc;
      s += acc; q += acc * acc;
    }
  }
  atomicAdd(&lsum[co], s);
  atomicAdd(&lssq[co], q);
  __syncthreads();
  if (tid < 128)       atomicAdd(&stats1[tid], lsum[tid]);
  else if (tid < 256)  atomicAdd(&stats1[tid], lssq[tid - 128]);
}

// ------------------------------ BN finalize --------------------------------
// sums: [sum(C) | sumsq(C)] -> st: [scale(C) | shift(C)]
__global__ void bn_finalize_kernel(const float* __restrict__ sums,
                                   const float* __restrict__ g,
                                   const float* __restrict__ bb,
                                   float* __restrict__ st, int C) {
  int c = threadIdx.x;
  if (c < C) {
    float mu  = sums[c] * INV_N;
    float var = sums[C + c] * INV_N - mu * mu;
    float s   = g[c] * rsqrtf(var + 1e-5f);
    st[c]     = s;
    st[C + c] = bb[c] - mu * s;
  }
}

// ------------------------------ conv2 (128 -> 256, k5) ----------------------
// Block = 256 thr (8 waves). Block tile: all 256 out channels x 64 positions.
// Wave w: M-subtiles {2w, 2w+1}, all 4 N-subtiles.  20 K-steps of 32 (5 taps).
__global__ void __launch_bounds__(256) conv2_wmma_kernel(
    const _Float16* __restrict__ y1t, const _Float16* __restrict__ wp2,
    const float* __restrict__ st1, _Float16* __restrict__ y2t,
    float* __restrict__ stats2) {
  __shared__ float ls[128], lt[128];
  __shared__ _Float16 xs[68 * 136];     // [row=pos-(p0-2)][ci], +8 pad
  __shared__ float lsum[256], lssq[256];
  const int tid  = threadIdx.x;
  const int tile = blockIdx.x;          // 65 tiles of 64 positions
  const int b    = blockIdx.y;
  const int p0   = tile * 64;
  if (tid < 128) { ls[tid] = st1[tid]; lt[tid] = st1[128 + tid]; }
  lsum[tid] = 0.f; lssq[tid] = 0.f;
  __syncthreads();
  // Stage input tile (BN1 affine + ReLU applied on the fly), zero halo.
  for (int qd = tid; qd < 68 * 32; qd += 256) {
    int row = qd >> 5;
    int c4  = (qd & 31) * 4;
    int p   = p0 - 2 + row;
    v4h out;
    if (p >= 0 && p < LOUT) {
      v4h src = *(const v4h*)(y1t + ((size_t)b * LOUT + p) * 128 + c4);
#pragma unroll
      for (int i = 0; i < 4; i++) {
        float v = ls[c4 + i] * (float)src[i] + lt[c4 + i];
        out[i] = (_Float16)(v > 0.f ? v : 0.f);
      }
    } else {
      out = (v4h)(_Float16)0.f;
    }
    *(v4h*)(&xs[row * 136 + c4]) = out;
  }
  __syncthreads();

  const int wave = tid >> 5, lane = tid & 31;
  const int half = lane >> 4, ln = lane & 15;
  v8f acc[2][4] = {};
  for (int kk = 0; kk < 5; kk++) {
    for (int ch = 0; ch < 4; ch++) {
      const _Float16* wpa = wp2 + (size_t)((((kk * 4 + ch) * 16 + wave * 2) * 32) + lane) * 16;
      v16h a0 = *(const v16h*)wpa;
      v16h a1 = *(const v16h*)(wpa + 32 * 16);
#pragma unroll
      for (int nt = 0; nt < 4; nt++) {
        int row = nt * 16 + ln + kk;
        const _Float16* bp = &xs[row * 136 + ch * 32 + half * 8];
        v8h blo = *(const v8h*)bp;
        v8h bhi = *(const v8h*)(bp + 16);
        v16h bf;
#pragma unroll
        for (int i = 0; i < 8; i++) { bf[i] = blo[i]; bf[8 + i] = bhi[i]; }
        acc[0][nt] = __builtin_amdgcn_wmma_f32_16x16x32_f16(
            false, a0, false, bf, (short)0, acc[0][nt], false, false);
        acc[1][nt] = __builtin_amdgcn_wmma_f32_16x16x32_f16(
            false, a1, false, bf, (short)0, acc[1][nt], false, false);
      }
    }
  }

  // Store f16 (pre-BN) + accumulate per-channel stats.
  float sm[2][8] = {}, sq[2][8] = {};
#pragma unroll
  for (int m = 0; m < 2; m++) {
    int cob = (wave * 2 + m) * 16 + half * 8;
#pragma unroll
    for (int nt = 0; nt < 4; nt++) {
      int pos = p0 + nt * 16 + ln;
      bool valid = pos < LOUT;
      v8h hv;
#pragma unroll
      for (int r = 0; r < 8; r++) {
        float v = acc[m][nt][r];
        hv[r] = (_Float16)v;
        if (valid) { sm[m][r] += v; sq[m][r] += v * v; }
      }
      if (valid) *(v8h*)(y2t + ((size_t)b * LOUT + pos) * 256 + cob) = hv;
    }
  }
#pragma unroll
  for (int m = 0; m < 2; m++) {
#pragma unroll
    for (int r = 0; r < 8; r++) {
      float s = sm[m][r], q = sq[m][r];
      for (int off = 1; off < 16; off <<= 1) {
        s += __shfl_xor(s, off, 32);
        q += __shfl_xor(q, off, 32);
      }
      if (ln == 0) {
        int cob = (wave * 2 + m) * 16 + half * 8;
        atomicAdd(&lsum[cob + r], s);
        atomicAdd(&lssq[cob + r], q);
      }
    }
  }
  __syncthreads();
  atomicAdd(&stats2[tid], lsum[tid]);
  atomicAdd(&stats2[256 + tid], lssq[tid]);
}

// ------------------------------ conv3 (256 -> 128, k3) ----------------------
__global__ void __launch_bounds__(256) conv3_wmma_kernel(
    const _Float16* __restrict__ y2t, const _Float16* __restrict__ wp3,
    const float* __restrict__ st2, _Float16* __restrict__ y3t,
    float* __restrict__ stats3) {
  __shared__ float ls[256], lt[256];
  __shared__ _Float16 xs[66 * 264];     // [row=pos-(p0-1)][ci 0..255], +8 pad
  __shared__ float lsum[128], lssq[128];
  const int tid  = threadIdx.x;
  const int tile = blockIdx.x;
  const int b    = blockIdx.y;
  const int p0   = tile * 64;
  ls[tid] = st2[tid]; lt[tid] = st2[256 + tid];
  if (tid < 128) { lsum[tid] = 0.f; lssq[tid] = 0.f; }
  __syncthreads();
  for (int qd = tid; qd < 66 * 64; qd += 256) {
    int row = qd >> 6;
    int c4  = (qd & 63) * 4;
    int p   = p0 - 1 + row;
    v4h out;
    if (p >= 0 && p < LOUT) {
      v4h src = *(const v4h*)(y2t + ((size_t)b * LOUT + p) * 256 + c4);
#pragma unroll
      for (int i = 0; i < 4; i++) {
        float v = ls[c4 + i] * (float)src[i] + lt[c4 + i];
        out[i] = (_Float16)(v > 0.f ? v : 0.f);
      }
    } else {
      out = (v4h)(_Float16)0.f;
    }
    *(v4h*)(&xs[row * 264 + c4]) = out;
  }
  __syncthreads();

  const int wave = tid >> 5, lane = tid & 31;
  const int half = lane >> 4, ln = lane & 15;
  v8f acc[4] = {};
  for (int kk = 0; kk < 3; kk++) {
    for (int ch = 0; ch < 8; ch++) {
      const _Float16* wpa = wp3 + (size_t)((((kk * 8 + ch) * 8 + wave) * 32) + lane) * 16;
      v16h a0 = *(const v16h*)wpa;
#pragma unroll
      for (int nt = 0; nt < 4; nt++) {
        int row = nt * 16 + ln + kk;
        const _Float16* bp = &xs[row * 264 + ch * 32 + half * 8];
        v8h blo = *(const v8h*)bp;
        v8h bhi = *(const v8h*)(bp + 16);
        v16h bf;
#pragma unroll
        for (int i = 0; i < 8; i++) { bf[i] = blo[i]; bf[8 + i] = bhi[i]; }
        acc[nt] = __builtin_amdgcn_wmma_f32_16x16x32_f16(
            false, a0, false, bf, (short)0, acc[nt], false, false);
      }
    }
  }

  float sm[8] = {}, sq[8] = {};
  const int cob = wave * 16 + half * 8;
#pragma unroll
  for (int nt = 0; nt < 4; nt++) {
    int pos = p0 + nt * 16 + ln;
    bool valid = pos < LOUT;
    v8h hv;
#pragma unroll
    for (int r = 0; r < 8; r++) {
      float v = acc[nt][r];
      hv[r] = (_Float16)v;
      if (valid) { sm[r] += v; sq[r] += v * v; }
    }
    if (valid) *(v8h*)(y3t + ((size_t)b * LOUT + pos) * 128 + cob) = hv;
  }
#pragma unroll
  for (int r = 0; r < 8; r++) {
    float s = sm[r], q = sq[r];
    for (int off = 1; off < 16; off <<= 1) {
      s += __shfl_xor(s, off, 32);
      q += __shfl_xor(q, off, 32);
    }
    if (ln == 0) { atomicAdd(&lsum[cob + r], s); atomicAdd(&lssq[cob + r], q); }
  }
  __syncthreads();
  if (tid < 128) {
    atomicAdd(&stats3[tid], lsum[tid]);
    atomicAdd(&stats3[128 + tid], lssq[tid]);
  }
}

// ------------------------- mean over length (BN3+ReLU) ----------------------
__global__ void __launch_bounds__(256) feat_reduce_kernel(
    const _Float16* __restrict__ y3t, const float* __restrict__ st3,
    float* __restrict__ feat) {
  const int b = blockIdx.x, tid = threadIdx.x;
  const int co = tid & 127, g = tid >> 7;
  __shared__ float red[256];
  const float s = st3[co], t = st3[128 + co];
  float acc = 0.f;
  for (int p = g; p < LOUT; p += 2) {
    float v = s * (float)y3t[((size_t)b * LOUT + p) * 128 + co] + t;
    acc += v > 0.f ? v : 0.f;
  }
  red[tid] = acc;
  __syncthreads();
  if (tid < 128) feat[b * 128 + co] = (red[co] + red[128 + co]) * (1.0f / 4097.0f);
}

// ------------------------------ NALU layers --------------------------------
__global__ void nalu1_kernel(const float* __restrict__ feat,
                             const float* __restrict__ What,
                             const float* __restrict__ Mhat,
                             const float* __restrict__ G,
                             float* __restrict__ h1) {
  const int i = blockIdx.x, j = threadIdx.x;   // 128 threads
  __shared__ float fr[128], lr[128];
  float v = feat[i * 128 + j];
  fr[j] = v;
  lr[j] = logf(fabsf(v) + 1e-10f);
  __syncthreads();
  float aa = 0.f, mm = 0.f, gg = 0.f;
  for (int k = 0; k < 128; k++) {
    float w = tanhf(What[j * 128 + k]) * (1.f / (1.f + expf(-Mhat[j * 128 + k])));
    aa += fr[k] * w;
    mm += lr[k] * w;
    gg += fr[k] * G[j * 128 + k];
  }
  float gate = 1.f / (1.f + expf(-gg));
  h1[i * 128 + j] = gate * aa + (1.f - gate) * expf(mm);
}

__global__ void nalu2_final_kernel(const float* __restrict__ h1,
                                   const float* __restrict__ What,
                                   const float* __restrict__ Mhat,
                                   const float* __restrict__ G,
                                   const float* __restrict__ fw,
                                   const float* __restrict__ fb,
                                   float* __restrict__ out) {
  const int i = blockIdx.x, tid = threadIdx.x;  // 32 threads
  __shared__ float fr[128], lr[128], h2[16];
  for (int k = tid; k < 128; k += 32) {
    float v = h1[i * 128 + k];
    fr[k] = v;
    lr[k] = logf(fabsf(v) + 1e-10f);
  }
  __syncthreads();
  if (tid < 16) {
    float aa = 0.f, mm = 0.f, gg = 0.f;
    for (int k = 0; k < 128; k++) {
      float w = tanhf(What[tid * 128 + k]) * (1.f / (1.f + expf(-Mhat[tid * 128 + k])));
      aa += fr[k] * w;
      mm += lr[k] * w;
      gg += fr[k] * G[tid * 128 + k];
    }
    float gate = 1.f / (1.f + expf(-gg));
    h2[tid] = gate * aa + (1.f - gate) * expf(mm);
  }
  __syncthreads();
  if (tid == 0) {
    float o = fb[0];
    for (int k = 0; k < 16; k++) o += h2[k] * fw[k];
    out[i] = o;
  }
}

// ------------------------------ host launcher ------------------------------
extern "C" void kernel_launch(void* const* d_in, const int* in_sizes, int n_in,
                              void* d_out, int out_size, void* d_ws, size_t ws_size,
                              hipStream_t stream) {
  (void)in_sizes; (void)n_in; (void)out_size; (void)ws_size;
  const float* x   = (const float*)d_in[0];
  const float* w1  = (const float*)d_in[1];
  const float* g1  = (const float*)d_in[3];
  const float* bb1 = (const float*)d_in[4];
  const float* w2  = (const float*)d_in[5];
  const float* g2  = (const float*)d_in[7];
  const float* bb2 = (const float*)d_in[8];
  const float* w3  = (const float*)d_in[9];
  const float* g3  = (const float*)d_in[11];
  const float* bb3 = (const float*)d_in[12];
  const float* n1W = (const float*)d_in[13];
  const float* n1M = (const float*)d_in[14];
  const float* n1G = (const float*)d_in[15];
  const float* n2W = (const float*)d_in[16];
  const float* n2M = (const float*)d_in[17];
  const float* n2G = (const float*)d_in[18];
  const float* fw  = (const float*)d_in[19];
  const float* fb  = (const float*)d_in[20];

  // Workspace layout (bytes, 256-aligned)
  const size_t Y1_BYTES = (size_t)B_ * LOUT * 128 * 2;   // 134,250,496
  const size_t Y2_BYTES = (size_t)B_ * LOUT * 256 * 2;   // 268,500,992
  char* ws = (char*)d_ws;
  _Float16* y1t   = (_Float16*)(ws);                     // reused as y3t
  _Float16* y2t   = (_Float16*)(ws + Y1_BYTES);
  _Float16* wp2   = (_Float16*)(ws + Y1_BYTES + Y2_BYTES);
  _Float16* wp3   = (_Float16*)(ws + Y1_BYTES + Y2_BYTES + 327680);
  float*    stats = (float*)   (ws + Y1_BYTES + Y2_BYTES + 327680 + 196608);
  float*    st    = stats + 1024;                        // [st1 256][st2 512][st3 256]
  float*    feat  = st + 1024;
  float*    h1    = feat + 128 * 128;
  _Float16* y3t   = y1t;
  float* stats1 = stats;        // [sum128|ssq128]
  float* stats2 = stats + 256;  // [sum256|ssq256]
  float* stats3 = stats + 768;  // [sum128|ssq128]
  float* st1 = st;
  float* st2 = st + 256;
  float* st3 = st + 768;

  hipMemsetAsync(stats, 0, 1024 * sizeof(float), stream);

  pack_w2_kernel<<<640, 256, 0, stream>>>(w2, wp2);
  pack_w3_kernel<<<384, 256, 0, stream>>>(w3, wp3);

  conv1_kernel<<<dim3(33, 128), 256, 0, stream>>>(x, w1, y1t, stats1);
  bn_finalize_kernel<<<1, 256, 0, stream>>>(stats1, g1, bb1, st1, 128);

  conv2_wmma_kernel<<<dim3(65, 128), 256, 0, stream>>>(y1t, wp2, st1, y2t, stats2);
  bn_finalize_kernel<<<1, 256, 0, stream>>>(stats2, g2, bb2, st2, 256);

  conv3_wmma_kernel<<<dim3(65, 128), 256, 0, stream>>>(y2t, wp3, st2, y3t, stats3);
  bn_finalize_kernel<<<1, 256, 0, stream>>>(stats3, g3, bb3, st3, 128);

  feat_reduce_kernel<<<128, 256, 0, stream>>>(y3t, st3, feat);
  nalu1_kernel<<<128, 128, 0, stream>>>(feat, n1W, n1M, n1G, h1);
  nalu2_final_kernel<<<128, 32, 0, stream>>>(h1, n2W, n2M, n2G, fw, fb, (float*)d_out);
}
